// CachedMultiHeadAttention_30700426231969
// MI455X (gfx1250) — matrix operand order repaired and checked by reading, not compiled, not run on hardware
//
#include <hip/hip_runtime.h>
#include <hip/hip_bf16.h>

typedef __attribute__((ext_vector_type(16))) __bf16 v16bf;
typedef __attribute__((ext_vector_type(8)))  float  v8f;

#define B_   16
#define Q_   32
#define KV_  4096
#define D_   1024
#define H_   16
#define DH_  64
#define NSPLIT 17          // 16 splits x 256 rows cover the 4096 cached rows, split 16 = the 32 current rows
#define ROWS_ (B_*Q_)      // 512

// D = A*B + C,  A 16x32 bf16, B 32x16 bf16, C/D 16x16 f32  (wave32)
__device__ __forceinline__ v8f wmma_bf16(v16bf a, v16bf b, v8f c) {
  return __builtin_amdgcn_wmma_f32_16x16x32_bf16(false, a, false, b, (short)0, c, false, false);
}

// 16-bit A-matrix (16x32) element->K map: lanes 0-15 hold K {0-7,16-23}, lanes 16-31 hold K {8-15,24-31}
__device__ __forceinline__ int a_kmap(int j, int hi) {
  return (hi ? 8 : 0) + j + ((j >= 8) ? 8 : 0);
}

// Generic pointers to __shared__ carry the LDS byte address in their low 32 bits
// (flat->LDS mapping truncates to addr[31:0]); that is what async-load VDST wants.
__device__ __forceinline__ unsigned lds_addr(const void* p) {
  return (unsigned)(size_t)p;
}

// Async DMA of 16 bytes/lane from global memory into LDS (tracked by ASYNCcnt).
__device__ __forceinline__ void async_ld_b128(unsigned lds_byte, const float* g) {
  asm volatile("global_load_async_to_lds_b128 %0, %1, off"
               :: "v"(lds_byte), "v"((unsigned long long)(size_t)g)
               : "memory");
}
__device__ __forceinline__ void wait_async_le32() {
  asm volatile("s_wait_asynccnt 32" ::: "memory");
}
__device__ __forceinline__ void wait_async_0() {
  asm volatile("s_wait_asynccnt 0" ::: "memory");
}

// ---------------------------------------------------------------------------
// Generic 512x1024x1024 GEMM:  Out[r,c] = (sum_k X[r,k]*W[k,c] + bias[c]) * scale
// 8 waves/block, one 16x16 tile per wave, bf16 WMMA with fp32 accumulation.
// ---------------------------------------------------------------------------
__global__ __launch_bounds__(256) void gemm512(const float* __restrict__ X,
                                               const float* __restrict__ W,
                                               const float* __restrict__ bias,
                                               float* __restrict__ Out,
                                               float scale)
{
  const int lane = threadIdx.x & 31;
  const int wave = threadIdx.x >> 5;
  const int tile = blockIdx.x * 8 + wave;   // 2048 tiles = 32 x 64
  const int mt = tile >> 6;
  const int nt = tile & 63;
  const int l15 = lane & 15;
  const int hi  = (lane & 16) ? 1 : 0;
  const int row = mt * 16 + l15;            // A-matrix row = lane&15
  const int col = nt * 16 + l15;            // B/C/D column = lane&15

  v8f c = {0.f,0.f,0.f,0.f,0.f,0.f,0.f,0.f};
  const float* xr = X + (long)row * D_;
  for (int k0 = 0; k0 < D_; k0 += 32) {
    v16bf a, b;
#pragma unroll
    for (int j = 0; j < 16; ++j)
      a[j] = (__bf16)xr[k0 + hi * 16 + j];
    const int kb = k0 + (hi ? 16 : 0);      // B rows: lanes 0-15 K=k0..k0+15, lanes 16-31 K=k0+16..k0+31
#pragma unroll
    for (int j = 0; j < 16; ++j)
      b[j] = (__bf16)W[(long)(kb + j) * D_ + col];
    c = wmma_bf16(a, b, c);
  }
  const float bb = bias ? bias[col] : 0.f;
#pragma unroll
  for (int r = 0; r < 8; ++r) {
    const int orow = mt * 16 + r + (hi ? 8 : 0);  // C/D row map
    Out[(long)orow * D_ + col] = (c[r] + bb) * scale;
  }
}

// ---------------------------------------------------------------------------
// Flash-decoding attention: one wave per (b, h, split).
// Splits 0..15 each stream 256 cached KV rows (8 steps of 32); split 16 does
// the 32 freshly-projected rows. K/V tiles are double-buffered through LDS
// with GLOBAL_LOAD_ASYNC_TO_LDS_B128 (ASYNCcnt), overlapping the next tile's
// HBM fetch with this tile's 16 WMMAs. Online softmax per wave; partial
// (m,l,acc) written to workspace and merged by attn_combine.
// ---------------------------------------------------------------------------
__global__ __launch_bounds__(32) void attn_split(const float* __restrict__ cache_k,
                                                 const float* __restrict__ cache_v,
                                                 const float* __restrict__ qs,   // q, pre-scaled by DH^-0.5
                                                 const float* __restrict__ ck,   // current K [512, D]
                                                 const float* __restrict__ cvv,  // current V [512, D]
                                                 float* __restrict__ pacc,
                                                 float* __restrict__ pm,
                                                 float* __restrict__ pl)
{
  __shared__ __align__(16) float KtileF[2][32 * 64];   // 2 x 8KB ping-pong K tile (fp32)
  __shared__ __align__(16) float VtileF[2][32 * 64];   // 2 x 8KB ping-pong V tile (fp32)
  __shared__ __bf16 Plds[32 * 32];                     // probs staging (C-layout -> A-layout)

  const int lane  = threadIdx.x;
  const int split = (int)(blockIdx.x % NSPLIT);
  const int bh    = (int)(blockIdx.x / NSPLIT);
  const int h     = bh & (H_ - 1);
  const int b     = bh / H_;
  const int l15   = lane & 15;
  const int hi    = (lane & 16) ? 1 : 0;

  // ---- q fragments: 32x64 -> A tiles [mtile][ktile] -------------------------
  v16bf aq[2][2];
  {
    const float* qp = qs + ((long)b * Q_ + l15) * D_ + h * DH_;
#pragma unroll
    for (int mtile = 0; mtile < 2; ++mtile) {
      const float* qrow = qp + (long)mtile * 16 * D_;
#pragma unroll
      for (int kt = 0; kt < 2; ++kt)
#pragma unroll
        for (int j = 0; j < 16; ++j)
          aq[mtile][kt][j] = (__bf16)qrow[kt * 32 + hi * 16 + j];
    }
  }

  float m_s[16], l_s[16];
#pragma unroll
  for (int i = 0; i < 16; ++i) { m_s[i] = -1e30f; l_s[i] = 0.f; }
  v8f acc[2][4];
#pragma unroll
  for (int mtile = 0; mtile < 2; ++mtile)
#pragma unroll
    for (int dt = 0; dt < 4; ++dt)
#pragma unroll
      for (int e = 0; e < 8; ++e) acc[mtile][dt][e] = 0.f;

  const int  nsteps = (split == 16) ? 1 : 8;
  const long sbase  = (split == 16) ? 0 : (long)split * 256;
  const float* Ks = (split == 16) ? (ck  + (long)b * Q_ * D_) : (cache_k + (long)b * KV_ * D_);
  const float* Vs = (split == 16) ? (cvv + (long)b * Q_ * D_) : (cache_v + (long)b * KV_ * D_);

  // Issue one 8KB 32x64 tile as 16 async b128 loads (2 coalesced 256B rows per op).
  auto issue_tile = [&](const float* src, long s0, unsigned lds_base) {
#pragma unroll
    for (int i = 0; i < 16; ++i) {
      const int f  = i * 32 + lane;        // float4 index within the tile
      const int r  = f >> 4;               // tile row (16 float4 per row)
      const int c4 = f & 15;               // float4 within row
      async_ld_b128(lds_base + (unsigned)f * 16u,
                    src + (s0 + r) * D_ + h * DH_ + c4 * 4);
    }
  };

  // prologue: stage step 0 into buffer 0
  int buf = 0;
  issue_tile(Ks, sbase, lds_addr(&KtileF[0][0]));
  issue_tile(Vs, sbase, lds_addr(&VtileF[0][0]));

  for (int step = 0; step < nsteps; ++step) {
    const long s0 = sbase + (long)step * 32;
    if (step + 1 < nsteps) {
      // stage next tile into the other buffer, then wait for the current one
      issue_tile(Ks, s0 + 32, lds_addr(&KtileF[buf ^ 1][0]));
      issue_tile(Vs, s0 + 32, lds_addr(&VtileF[buf ^ 1][0]));
      wait_async_le32();                   // >32 outstanding are the new ones; current buffer done
    } else {
      wait_async_0();
    }
    const float* Kt = KtileF[buf];
    const float* Vt = VtileF[buf];

    // K B-fragments from LDS: B[K=d(32)][N=s(16)], 64B contiguous per lane
    v16bf bk[2][2];
#pragma unroll
    for (int st = 0; st < 2; ++st) {
      const float* kr = Kt + (st * 16 + l15) * 64 + hi * 16;
#pragma unroll
      for (int kt = 0; kt < 2; ++kt)
#pragma unroll
        for (int j = 0; j < 16; ++j)
          bk[st][kt][j] = (__bf16)kr[kt * 32 + j];
    }

    // scores [32 rows x 32 s]: 8 wmma
    v8f sc[2][2];
#pragma unroll
    for (int mtile = 0; mtile < 2; ++mtile)
#pragma unroll
      for (int st = 0; st < 2; ++st) {
        v8f z = {0.f,0.f,0.f,0.f,0.f,0.f,0.f,0.f};
        z = wmma_bf16(aq[mtile][0], bk[st][0], z);
        sc[mtile][st] = wmma_bf16(aq[mtile][1], bk[st][1], z);
      }

    // online softmax over this 32-key step (row values live in one 16-lane half)
#pragma unroll
    for (int mtile = 0; mtile < 2; ++mtile)
#pragma unroll
      for (int e = 0; e < 8; ++e) {
        float x0 = sc[mtile][0][e], x1 = sc[mtile][1][e];
        float vmax = fmaxf(x0, x1);
#pragma unroll
        for (int off = 8; off >= 1; off >>= 1)
          vmax = fmaxf(vmax, __shfl_xor(vmax, off, 32));
        const int idx = mtile * 8 + e;
        const float mnew  = fmaxf(m_s[idx], vmax);
        const float alpha = __expf(m_s[idx] - mnew);
        const float p0 = __expf(x0 - mnew);
        const float p1 = __expf(x1 - mnew);
        float rs = p0 + p1;
#pragma unroll
        for (int off = 8; off >= 1; off >>= 1)
          rs += __shfl_xor(rs, off, 32);
        l_s[idx] = l_s[idx] * alpha + rs;
        m_s[idx] = mnew;
#pragma unroll
        for (int dt = 0; dt < 4; ++dt) acc[mtile][dt][e] *= alpha;
        const int prow = mtile * 16 + e + hi * 8;           // C-layout row
        Plds[prow * 32 + l15]      = (__bf16)p0;
        Plds[prow * 32 + 16 + l15] = (__bf16)p1;
      }

    // transpose P through LDS into A-layout fragments (same-wave DS ops are in order)
    v16bf ap[2];
#pragma unroll
    for (int mtile = 0; mtile < 2; ++mtile)
#pragma unroll
      for (int j = 0; j < 16; ++j)
        ap[mtile][j] = Plds[(mtile * 16 + l15) * 32 + hi * 16 + j];

    // V B-fragments from LDS: for fixed j all 16 lanes of a half read one row, consecutive banks
    v16bf bvf[4];
#pragma unroll
    for (int j = 0; j < 16; ++j) {
      const float* vrow = Vt + (hi * 16 + j) * 64;
#pragma unroll
      for (int dt = 0; dt < 4; ++dt)
        bvf[dt][j] = (__bf16)vrow[dt * 16 + l15];
    }

    // acc += P * V : 8 wmma
#pragma unroll
    for (int mtile = 0; mtile < 2; ++mtile)
#pragma unroll
      for (int dt = 0; dt < 4; ++dt)
        acc[mtile][dt] = wmma_bf16(ap[mtile], bvf[dt], acc[mtile][dt]);

    buf ^= 1;
  }

  // write partials (m, l, acc) for the combine pass
  const long base = ((long)bh * NSPLIT + split) * 32;
#pragma unroll
  for (int mtile = 0; mtile < 2; ++mtile)
#pragma unroll
    for (int e = 0; e < 8; ++e) {
      const int row = mtile * 16 + e + hi * 8;
      const int idx = mtile * 8 + e;
      if (l15 == 0) { pm[base + row] = m_s[idx]; pl[base + row] = l_s[idx]; }
#pragma unroll
      for (int dt = 0; dt < 4; ++dt)
        pacc[(base + row) * 64 + dt * 16 + l15] = acc[mtile][dt][e];
    }
}

// ---------------------------------------------------------------------------
// Merge the 17 split partials:  ctx = sum_i w_i*acc_i / sum_i w_i*l_i,
// w_i = exp(m_i - max_i m_i). One thread per (b,h,row,d).
// ---------------------------------------------------------------------------
__global__ __launch_bounds__(256) void attn_combine(const float* __restrict__ pacc,
                                                    const float* __restrict__ pm,
                                                    const float* __restrict__ pl,
                                                    float* __restrict__ ctx)
{
  const int t   = blockIdx.x * 256 + threadIdx.x;  // 16*16*32*64 = 524288
  const int d   = t & 63;
  const int row = (t >> 6) & 31;
  const int bh  = t >> 11;
  const int b   = bh >> 4;
  const int h   = bh & 15;

  float M = -1e30f;
#pragma unroll
  for (int i = 0; i < NSPLIT; ++i)
    M = fmaxf(M, pm[((long)bh * NSPLIT + i) * 32 + row]);
  float L = 0.f, A = 0.f;
#pragma unroll
  for (int i = 0; i < NSPLIT; ++i) {
    const long pb = ((long)bh * NSPLIT + i) * 32 + row;
    const float w = __expf(pm[pb] - M);
    L += w * pl[pb];
    A += w * pacc[pb * 64 + d];
  }
  ctx[((long)b * Q_ + row) * D_ + h * DH_ + d] = A / (L + 1e-30f);
}

// ---------------------------------------------------------------------------
extern "C" void kernel_launch(void* const* d_in, const int* in_sizes, int n_in,
                              void* d_out, int out_size, void* d_ws, size_t ws_size,
                              hipStream_t stream) {
  const float* x       = (const float*)d_in[0];
  const float* cache_k = (const float*)d_in[1];
  const float* cache_v = (const float*)d_in[2];
  const float* Wq      = (const float*)d_in[3];
  const float* bq      = (const float*)d_in[4];
  const float* Wk      = (const float*)d_in[5];
  const float* Wv      = (const float*)d_in[6];
  const float* bv      = (const float*)d_in[7];
  const float* Wo      = (const float*)d_in[8];
  const float* bo      = (const float*)d_in[9];
  float* out = (float*)d_out;

  // workspace carve-out (floats): q_s | ck | cv | ctx | pacc | pm | pl  (~45 MB)
  float* ws   = (float*)d_ws;
  float* q_s  = ws;
  float* ck   = q_s  + (size_t)ROWS_ * D_;
  float* cv   = ck   + (size_t)ROWS_ * D_;
  float* ctx  = cv   + (size_t)ROWS_ * D_;
  float* pacc = ctx  + (size_t)ROWS_ * D_;
  float* pm   = pacc + (size_t)B_ * H_ * NSPLIT * Q_ * DH_;
  float* pl   = pm   + (size_t)B_ * H_ * NSPLIT * Q_;

  // Both DH^-0.25 scales (on q and on k) fold into one DH^-0.5 = 1/8 on q (incl. bias).
  const float qscale = 0.125f;

  gemm512<<<256, 256, 0, stream>>>(x, Wq, bq,      q_s, qscale);
  gemm512<<<256, 256, 0, stream>>>(x, Wk, nullptr, ck,  1.0f);
  gemm512<<<256, 256, 0, stream>>>(x, Wv, bv,      cv,  1.0f);

  attn_split<<<B_ * H_ * NSPLIT, 32, 0, stream>>>(cache_k, cache_v, q_s, ck, cv, pacc, pm, pl);
  attn_combine<<<(B_ * H_ * Q_ * DH_) / 256, 256, 0, stream>>>(pacc, pm, pl, ctx);

  gemm512<<<256, 256, 0, stream>>>(ctx, Wo, bo, out, 1.0f);
}
